// Graphormer_53575422050474
// MI455X (gfx1250) — matrix operand order repaired and checked by reading, not compiled, not run on hardware
//
#include <hip/hip_runtime.h>
#include <hip/hip_bf16.h>
#include <math.h>

// ===========================================================================
// Graphormer forward for MI455X (gfx1250, wave32, WMMA).
// Roofline: 2 x 537MB spatial tables re-read per layer -> 4.3GB HBM traffic
// (~185us floor @ 23.3TB/s); ~13 GFLOP of GEMM is noise at WMMA rates.
// GEMMs: v_wmma_f32_16x16x32_bf16; weights pre-converted to bf16 and
// pre-transposed so LDS tiles stage as pure b128 copies -> async global->LDS
// (ASYNCcnt) with double buffering when the toolchain exposes the builtins.
// ===========================================================================

typedef __attribute__((ext_vector_type(16))) __bf16 bf16x16;
typedef __attribute__((ext_vector_type(8)))  float  f32x8;
typedef __attribute__((ext_vector_type(4)))  int    v4i;

#define N_TOK  512
#define H_DIM  512
#define NHEAD  8
#define DH     64
#define FF_DIM 2048
#define LAYERS 4

static __device__ __forceinline__ __bf16 tobf(float f) { return (__bf16)f; }

union FragU {            // WMMA 16x32 bf16 fragment = two aligned b128 LDS reads
    bf16x16 v;
    uint4   q[2];
};

// ---- CDNA5 async global->LDS copy (guarded: falls back to plain copy) -----
#if defined(__AMDGCN__) && \
    __has_builtin(__builtin_amdgcn_global_load_async_to_lds_b128) && \
    __has_builtin(__builtin_amdgcn_s_wait_asynccnt)
#define USE_ASYNC_LDS 1
#else
#define USE_ASYNC_LDS 0
#endif

typedef __attribute__((address_space(1))) v4i gbl_v4i;   // global int4
typedef __attribute__((address_space(3))) v4i lds_v4i;   // LDS int4

static __device__ __forceinline__ lds_v4i* to_lds(void* p) {
    // generic->LDS addrspacecast == low-32 truncation on AMDGPU
    return (lds_v4i*)(unsigned int)(unsigned long long)p;
}
static __device__ __forceinline__ gbl_v4i* to_gbl(const void* p) {
    return (gbl_v4i*)(unsigned long long)p;
}

static __device__ __forceinline__ void copy16_g2l(const void* g, void* l) {
#if USE_ASYNC_LDS
    __builtin_amdgcn_global_load_async_to_lds_b128(to_gbl(g), to_lds(l), 0, 0);
#else
    *(uint4*)l = *(const uint4*)g;
#endif
}
static __device__ __forceinline__ void async_wait_all() {
#if USE_ASYNC_LDS
    __builtin_amdgcn_s_wait_asynccnt(0);
#endif
}

// ---------------------------------------------------------------------------
// GEMM: C[M,N] = act(A[M,K] @ B[K,N] + bias) (+ residual)
//   A  : bf16 [M,K] row-major (pre-converted activations)
//   BT : bf16 [N,K] row-major (weights pre-transposed+converted)
//   Cf : optional f32 output; Cbf: optional bf16 output (for next GEMM's A)
// 256 threads = 8 waves; block tile 128x64; wave tile 16x64; K-step 64
// (8 static v_wmma / iter). Double-buffered LDS staged via async b128 copies.
// LDS row stride 72 bf16 = 144B (16B-aligned, bank-optimal for b128 reads).
// ---------------------------------------------------------------------------
__global__ __launch_bounds__(256)
void gemm_wmma_bf16(const __bf16* __restrict__ A, const __bf16* __restrict__ BT,
                    const float* __restrict__ bias, const float* __restrict__ resid,
                    float* __restrict__ Cf, __bf16* __restrict__ Cbf,
                    int M, int N, int K, int gelu)
{
    __shared__ __align__(16) __bf16 As[2][128][72];
    __shared__ __align__(16) __bf16 Bt[2][64][72];

    const int tid  = threadIdx.x;
    const int lane = tid & 31;
    const int wave = tid >> 5;
    const int m0   = blockIdx.y * 128;
    const int n0   = blockIdx.x * 64;

    f32x8 acc[4];
    #pragma unroll
    for (int c = 0; c < 4; ++c)
        #pragma unroll
        for (int r = 0; r < 8; ++r) acc[c][r] = 0.0f;

    // Per-lane fragment addressing (ISA 7.12.2, 16-bit operands, wave32):
    //  A (16x32): lane&15 = row m; lanes 16-31 use K-base +8 (bytes +16).
    //  B (32x16): lane&15 = col n; lanes 16-31 use K-base +8 (bytes +16).
    const int am  = (wave << 4) + (lane & 15);
    const int kb2 = (lane & 16) ? 16 : 0;
    const int kr2 = (lane & 16) ? 16 : 0;

    auto stage = [&](int buf, int k0) {
        #pragma unroll
        for (int t = 0; t < 4; ++t) {           // A tile: 128 rows x 128B
            const int f = tid + t * 256, row = f >> 3, c = (f & 7) << 3;
            copy16_g2l(&A[(size_t)(m0 + row) * K + k0 + c], &As[buf][row][c]);
        }
        #pragma unroll
        for (int t = 0; t < 2; ++t) {           // B tile: 64 rows x 128B
            const int f = tid + t * 256, row = f >> 3, c = (f & 7) << 3;
            copy16_g2l(&BT[(size_t)(n0 + row) * K + k0 + c], &Bt[buf][row][c]);
        }
    };

    stage(0, 0);
    for (int k0 = 0; k0 < K; k0 += 64) {
        const int cur = (k0 >> 6) & 1;
        async_wait_all();                        // my copies into `cur` landed
        __syncthreads();                         // everyone's landed; prev reads done
        if (k0 + 64 < K) stage(cur ^ 1, k0 + 64);  // overlap next tile copy

        const unsigned char* Abase = (const unsigned char*)&As[cur][0][0];
        const unsigned char* Bbase = (const unsigned char*)&Bt[cur][0][0];
        #pragma unroll
        for (int s = 0; s < 2; ++s) {            // two K=32 sub-steps
            FragU a;
            const unsigned char* ap = Abase + am * 144 + s * 64 + kb2;
            a.q[0] = *reinterpret_cast<const uint4*>(ap);
            a.q[1] = *reinterpret_cast<const uint4*>(ap + 32);
            #pragma unroll
            for (int nt = 0; nt < 4; ++nt) {
                FragU b;
                const int bn = (nt << 4) + (lane & 15);
                const unsigned char* bp = Bbase + bn * 144 + s * 64 + kr2;
                b.q[0] = *reinterpret_cast<const uint4*>(bp);
                b.q[1] = *reinterpret_cast<const uint4*>(bp + 32);
                acc[nt] = __builtin_amdgcn_wmma_f32_16x16x32_bf16(
                    false, a.v, false, b.v, (short)0, acc[nt], false, false);
            }
        }
    }

    // --- epilogue: f32 C layout (VGPR r: lanes0-15 M=r, lanes16-31 M=r+8) ---
    #pragma unroll
    for (int nt = 0; nt < 4; ++nt) {
        const int n = n0 + (nt << 4) + (lane & 15);
        const float bval = bias ? bias[n] : 0.0f;
        #pragma unroll
        for (int r = 0; r < 8; ++r) {
            const int m = m0 + (wave << 4) + r + ((lane & 16) ? 8 : 0);
            float v = acc[nt][r] + bval;
            if (gelu) v = 0.5f * v * (1.0f + erff(v * 0.70710678118654752f));
            if (resid) v += resid[(size_t)m * N + n];
            if (Cf)  Cf[(size_t)m * N + n]  = v;
            if (Cbf) Cbf[(size_t)m * N + n] = tobf(v);
        }
    }
}

// ---------------------------------------------------------------------------
// f32 -> bf16 elementwise convert (for the input x)
// ---------------------------------------------------------------------------
__global__ __launch_bounds__(256)
void cvt_bf16(const float* __restrict__ src, __bf16* __restrict__ dst, int n)
{
    const int i = (blockIdx.x * 256 + threadIdx.x) * 4;
    if (i + 3 < n) {
        const float4 v = *reinterpret_cast<const float4*>(src + i);
        dst[i + 0] = tobf(v.x);
        dst[i + 1] = tobf(v.y);
        dst[i + 2] = tobf(v.z);
        dst[i + 3] = tobf(v.w);
    }
}

// ---------------------------------------------------------------------------
// W[K,N] f32 -> Wt[N,K] bf16 (transpose + convert), blockIdx.z = matrix index
// ---------------------------------------------------------------------------
__global__ __launch_bounds__(256)
void transpose_cvt(const float* __restrict__ W, __bf16* __restrict__ Wt, int K, int N)
{
    __shared__ float tile[32][33];
    const int tid = threadIdx.x;
    const size_t off = (size_t)blockIdx.z * K * N;
    const int nb = blockIdx.x * 32;
    const int kb = blockIdx.y * 32;
    #pragma unroll
    for (int t = 0; t < 4; ++t) {
        const int e = tid + t * 256, r = e >> 5, c = e & 31;
        tile[r][c] = W[off + (size_t)(kb + r) * N + nb + c];
    }
    __syncthreads();
    #pragma unroll
    for (int t = 0; t < 4; ++t) {
        const int e = tid + t * 256, r = e >> 5, c = e & 31;
        Wt[off + (size_t)(nb + r) * K + kb + c] = tobf(tile[c][r]);
    }
}

// ---------------------------------------------------------------------------
// LayerNorm over H=512, one row per block; writes f32 and/or bf16 output.
// ---------------------------------------------------------------------------
__global__ __launch_bounds__(256)
void layernorm512(const float* __restrict__ x, const float* __restrict__ g,
                  const float* __restrict__ b, float* __restrict__ yf,
                  __bf16* __restrict__ ybf)
{
    __shared__ float red[256];
    const int row = blockIdx.x;
    const int tid = threadIdx.x;
    const float v0 = x[(size_t)row * H_DIM + tid];
    const float v1 = x[(size_t)row * H_DIM + tid + 256];

    red[tid] = v0 + v1;
    __syncthreads();
    for (int s = 128; s > 0; s >>= 1) {
        if (tid < s) red[tid] += red[tid + s];
        __syncthreads();
    }
    const float mean = red[0] * (1.0f / H_DIM);
    __syncthreads();

    const float d0 = v0 - mean, d1 = v1 - mean;
    red[tid] = d0 * d0 + d1 * d1;
    __syncthreads();
    for (int s = 128; s > 0; s >>= 1) {
        if (tid < s) red[tid] += red[tid + s];
        __syncthreads();
    }
    const float rstd = rsqrtf(red[0] * (1.0f / H_DIM) + 1e-5f);

    const float o0 = d0 * rstd * g[tid]       + b[tid];
    const float o1 = d1 * rstd * g[tid + 256] + b[tid + 256];
    if (yf) {
        yf[(size_t)row * H_DIM + tid]       = o0;
        yf[(size_t)row * H_DIM + tid + 256] = o1;
    }
    if (ybf) {
        ybf[(size_t)row * H_DIM + tid]       = tobf(o0);
        ybf[(size_t)row * H_DIM + tid + 256] = tobf(o1);
    }
}

// ---------------------------------------------------------------------------
// Fused attention row: one block per (i, h). Streams the two 537MB spatial
// tables exactly once per layer (b128 loads + global_prefetch_b8), fuses
// q.k + q.SPQ + k.SPK, bias, softmax, p@V. Output written as bf16 (it only
// feeds the Wo GEMM). Q/K/V stored [N, H].
// ---------------------------------------------------------------------------
__global__ __launch_bounds__(256)
void attention_row(const float* __restrict__ Q, const float* __restrict__ Km,
                   const float* __restrict__ V, const float* __restrict__ spq,
                   const float* __restrict__ spk, const float* __restrict__ bias,
                   __bf16* __restrict__ O)
{
    __shared__ __align__(16) float qs[DH];
    __shared__ float sc[N_TOK];
    __shared__ float red[256];
    __shared__ float opart[4][DH];

    const int i = blockIdx.x;
    const int h = blockIdx.y;
    const int tid = threadIdx.x;

    if (tid < DH) qs[tid] = Q[(size_t)i * H_DIM + h * DH + tid];
    __syncthreads();

    const float scale = 0.125f;  // D^-0.5 = 64^-0.5
    float lmax = -3.0e38f;

    #pragma unroll
    for (int jj = 0; jj < 2; ++jj) {
        const int j = tid + jj * 256;
        const float4* kp = reinterpret_cast<const float4*>(&Km[(size_t)j * H_DIM + h * DH]);
        const size_t tbase = ((((size_t)h * N_TOK + i) * N_TOK) + j) * DH;
        const float4* sq = reinterpret_cast<const float4*>(&spq[tbase]);
        const float4* sk = reinterpret_cast<const float4*>(&spk[tbase]);
        const float4* qp = reinterpret_cast<const float4*>(qs);
        __builtin_prefetch((const void*)(sq + 16), 0, 1);
        __builtin_prefetch((const void*)(sk + 16), 0, 1);

        float s = 0.0f;
        #pragma unroll
        for (int d = 0; d < DH / 4; ++d) {
            const float4 kv = kp[d];
            const float4 aq = sq[d];
            const float4 ak = sk[d];
            const float4 qv = qp[d];
            s += qv.x * (kv.x + aq.x) + kv.x * ak.x;
            s += qv.y * (kv.y + aq.y) + kv.y * ak.y;
            s += qv.z * (kv.z + aq.z) + kv.z * ak.z;
            s += qv.w * (kv.w + aq.w) + kv.w * ak.w;
        }
        s = s * scale + bias[(((size_t)h * N_TOK) + i) * N_TOK + j];
        sc[j] = s;
        lmax = fmaxf(lmax, s);
    }

    red[tid] = lmax;
    __syncthreads();
    for (int s = 128; s > 0; s >>= 1) {
        if (tid < s) red[tid] = fmaxf(red[tid], red[tid + s]);
        __syncthreads();
    }
    const float mx = red[0];
    __syncthreads();

    float lsum = 0.0f;
    #pragma unroll
    for (int jj = 0; jj < 2; ++jj) {
        const int j = tid + jj * 256;
        const float p = __expf(sc[j] - mx);
        sc[j] = p;
        lsum += p;
    }
    red[tid] = lsum;
    __syncthreads();
    for (int s = 128; s > 0; s >>= 1) {
        if (tid < s) red[tid] += red[tid + s];
        __syncthreads();
    }
    const float inv = 1.0f / red[0];
    __syncthreads();

    const int d = tid & 63;
    const int chunk = tid >> 6;
    const float* vp = &V[(size_t)h * DH + d];
    float a = 0.0f;
    const int j0 = chunk * 128;
    for (int j = j0; j < j0 + 128; ++j)
        a += sc[j] * vp[(size_t)j * H_DIM];
    opart[chunk][d] = a;
    __syncthreads();
    if (tid < DH)
        O[(size_t)i * H_DIM + h * DH + tid] =
            tobf((opart[0][tid] + opart[1][tid] + opart[2][tid] + opart[3][tid]) * inv);
}

// ---------------------------------------------------------------------------
// Host orchestration
// ---------------------------------------------------------------------------
extern "C" void kernel_launch(void* const* d_in, const int* in_sizes, int n_in,
                              void* d_out, int out_size, void* d_ws, size_t ws_size,
                              hipStream_t stream)
{
    const float* x    = (const float*)d_in[0];
    const float* bias = (const float*)d_in[1];
    const float* spq  = (const float*)d_in[2];
    const float* spk  = (const float*)d_in[3];
    const float* Wat  = (const float*)d_in[4];
    const float* bat  = (const float*)d_in[5];
    const float* ln1g = (const float*)d_in[6];
    const float* ln1b = (const float*)d_in[7];
    const float* Wq   = (const float*)d_in[8];
    const float* bq   = (const float*)d_in[9];
    const float* Wk   = (const float*)d_in[10];
    const float* bk   = (const float*)d_in[11];
    const float* Wv   = (const float*)d_in[12];
    const float* bv   = (const float*)d_in[13];
    const float* Wo   = (const float*)d_in[14];
    const float* bo   = (const float*)d_in[15];
    const float* ln2g = (const float*)d_in[16];
    const float* ln2b = (const float*)d_in[17];
    const float* W1   = (const float*)d_in[18];
    const float* b1   = (const float*)d_in[19];
    const float* W2   = (const float*)d_in[20];
    const float* b2   = (const float*)d_in[21];
    const float* fg   = (const float*)d_in[22];
    const float* fb   = (const float*)d_in[23];

    char* w = (char*)d_ws;
    float*  hbuf = (float*)w;  w += (size_t)N_TOK * H_DIM * 4;
    float*  Qb   = (float*)w;  w += (size_t)N_TOK * H_DIM * 4;
    float*  Kb   = (float*)w;  w += (size_t)N_TOK * H_DIM * 4;
    float*  Vb   = (float*)w;  w += (size_t)N_TOK * H_DIM * 4;
    __bf16* xbf  = (__bf16*)w; w += (size_t)N_TOK * 128   * 2;
    __bf16* ybf  = (__bf16*)w; w += (size_t)N_TOK * H_DIM * 2;
    __bf16* Obf  = (__bf16*)w; w += (size_t)N_TOK * H_DIM * 2;
    __bf16* Fbf  = (__bf16*)w; w += (size_t)N_TOK * FF_DIM * 2;
    __bf16* WatT = (__bf16*)w; w += (size_t)128 * H_DIM * 2;
    __bf16* WqT  = (__bf16*)w; w += (size_t)LAYERS * H_DIM * H_DIM * 2;
    __bf16* WkT  = (__bf16*)w; w += (size_t)LAYERS * H_DIM * H_DIM * 2;
    __bf16* WvT  = (__bf16*)w; w += (size_t)LAYERS * H_DIM * H_DIM * 2;
    __bf16* WoT  = (__bf16*)w; w += (size_t)LAYERS * H_DIM * H_DIM * 2;
    __bf16* W1T  = (__bf16*)w; w += (size_t)LAYERS * H_DIM * FF_DIM * 2;
    __bf16* W2T  = (__bf16*)w; w += (size_t)LAYERS * FF_DIM * H_DIM * 2;

    const dim3 blk(256);
    const dim3 g512(H_DIM / 64, N_TOK / 128);      // (8,4)
    const dim3 gff (FF_DIM / 64, N_TOK / 128);     // (32,4)
    const dim3 gatt(N_TOK, NHEAD);                 // (512,8)

    // ---- one-time (per launch) conversions: x -> bf16, W -> bf16 transposed
    cvt_bf16<<<(N_TOK * 128) / 1024, blk, 0, stream>>>(x, xbf, N_TOK * 128);
    transpose_cvt<<<dim3(H_DIM / 32, 128 / 32, 1), blk, 0, stream>>>(Wat, WatT, 128, H_DIM);
    transpose_cvt<<<dim3(16, 16, LAYERS), blk, 0, stream>>>(Wq, WqT, H_DIM, H_DIM);
    transpose_cvt<<<dim3(16, 16, LAYERS), blk, 0, stream>>>(Wk, WkT, H_DIM, H_DIM);
    transpose_cvt<<<dim3(16, 16, LAYERS), blk, 0, stream>>>(Wv, WvT, H_DIM, H_DIM);
    transpose_cvt<<<dim3(16, 16, LAYERS), blk, 0, stream>>>(Wo, WoT, H_DIM, H_DIM);
    transpose_cvt<<<dim3(64, 16, LAYERS), blk, 0, stream>>>(W1, W1T, H_DIM, FF_DIM);
    transpose_cvt<<<dim3(16, 64, LAYERS), blk, 0, stream>>>(W2, W2T, FF_DIM, H_DIM);

    // atom encoder: h = x @ W_atom + b_atom   (M=512, N=512, K=128)
    gemm_wmma_bf16<<<g512, blk, 0, stream>>>(xbf, WatT, bat, nullptr,
                                             hbuf, nullptr, N_TOK, H_DIM, 128, 0);

    for (int l = 0; l < LAYERS; ++l) {
        const size_t wofs = (size_t)l * H_DIM * H_DIM;
        layernorm512<<<N_TOK, blk, 0, stream>>>(hbuf, ln1g + l * H_DIM, ln1b + l * H_DIM,
                                                nullptr, ybf);

        gemm_wmma_bf16<<<g512, blk, 0, stream>>>(ybf, WqT + wofs, bq + l * H_DIM,
                                                 nullptr, Qb, nullptr, N_TOK, H_DIM, H_DIM, 0);
        gemm_wmma_bf16<<<g512, blk, 0, stream>>>(ybf, WkT + wofs, bk + l * H_DIM,
                                                 nullptr, Kb, nullptr, N_TOK, H_DIM, H_DIM, 0);
        gemm_wmma_bf16<<<g512, blk, 0, stream>>>(ybf, WvT + wofs, bv + l * H_DIM,
                                                 nullptr, Vb, nullptr, N_TOK, H_DIM, H_DIM, 0);

        attention_row<<<gatt, blk, 0, stream>>>(Qb, Kb, Vb, spq, spk, bias, Obf);

        // h = h + O @ Wo + bo  (residual read/write same element by same thread)
        gemm_wmma_bf16<<<g512, blk, 0, stream>>>(Obf, WoT + wofs, bo + l * H_DIM,
                                                 hbuf, hbuf, nullptr, N_TOK, H_DIM, H_DIM, 0);

        layernorm512<<<N_TOK, blk, 0, stream>>>(hbuf, ln2g + l * H_DIM, ln2b + l * H_DIM,
                                                nullptr, ybf);

        gemm_wmma_bf16<<<gff, blk, 0, stream>>>(ybf, W1T + (size_t)l * H_DIM * FF_DIM,
                                                b1 + (size_t)l * FF_DIM, nullptr,
                                                nullptr, Fbf, N_TOK, FF_DIM, H_DIM, 1);
        gemm_wmma_bf16<<<g512, blk, 0, stream>>>(Fbf, W2T + (size_t)l * FF_DIM * H_DIM,
                                                 b2 + l * H_DIM, hbuf,
                                                 hbuf, nullptr, N_TOK, H_DIM, FF_DIM, 0);
    }

    layernorm512<<<N_TOK, blk, 0, stream>>>(hbuf, fg, fb, (float*)d_out, nullptr);
}
